// LSTM_54571854463460
// MI455X (gfx1250) — matrix operand order repaired and checked by reading, not compiled, not run on hardware
//
#include <hip/hip_runtime.h>
#include <hip/hip_bf16.h>

typedef __attribute__((ext_vector_type(16))) __bf16 v16bf;
typedef __attribute__((ext_vector_type(8)))  float  v8f;
typedef __attribute__((ext_vector_type(4)))  unsigned int v4u;
typedef __attribute__((ext_vector_type(8)))  int v8i;
typedef __attribute__((ext_vector_type(4)))  int v4i;

#define BB 256      // batch
#define TT 128      // timesteps
#define HH 1024     // hidden
#define CC 10       // classes
#define KTILE 32    // K per WMMA
#define NKT (HH / KTILE)     // 32 k-tiles
#define NMT (BB / 16)        // 16 m-tiles
#define CHUNK_KT 4           // k-tiles staged per TDM transfer
#define NCHUNK (NKT / CHUNK_KT)          // 8 chunks
#define CHUNK_ELEMS (CHUNK_KT * 4 * 512) // 8192 bf16 = 16KB per chunk
#define CHUNK_BYTES (CHUNK_ELEMS * 2)
#define WPACK_ELEMS (4 * HH * HH)        // packed recurrent weights (bf16)

// workspace layout (bytes)
//  [0, 8MB)      : packed Wh bf16; tile = ((nt*32+kt)*4 + g), 512 elems (1KB) each
//                  -> one (nt, 4-k-step) chunk of all 4 gates is 16KB contiguous
//  [8MB, 8.5MB)  : h bf16 buffer 0 (B*H)
//  [8.5MB, 9MB)  : h bf16 buffer 1 (B*H)
//  [9MB, 10MB)   : c fp32 (B*H)
//  [10MB, 11MB)  : h fp32 (B*H)
#define OFF_HBF  ((size_t)WPACK_ELEMS * 2)
#define OFF_C    (OFF_HBF + (size_t)2 * BB * HH * 2)
#define OFF_HF   (OFF_C + (size_t)BB * HH * 4)

#if __has_builtin(__builtin_amdgcn_tensor_load_to_lds) && \
    __has_builtin(__builtin_amdgcn_s_wait_tensorcnt)
#define HAVE_TDM 1
#else
#define HAVE_TDM 0
#endif

#if __has_builtin(__builtin_amdgcn_tanhf)
#define TANHF(x) __builtin_amdgcn_tanhf(x)
#else
#define TANHF(x) tanhf(x)
#endif

static __device__ __forceinline__ float sigf(float x) {
    return __builtin_amdgcn_rcpf(1.0f + __expf(-x));
}

static __device__ __forceinline__ unsigned short f2bf(float f) {
    unsigned int u = __float_as_uint(f);
    unsigned int r = (u + 0x7FFFu + ((u >> 16) & 1u)) >> 16;   // round-to-nearest-even
    return (unsigned short)r;
}

#if HAVE_TDM
// 1-D TDM copy: 16KB (4096 dwords) from global to LDS.  D# per ISA ch.8:
// group0: count=1 | lds_addr | global_addr(57b) | type=2
// group1: data_size=4B; tensor_dim0=tile_dim0=4096 dwords; stride0=4096; rest 0
// (this toolchain's builtin is the 6-arg form: g0, g1, g2, g3, extra, cpol)
static __device__ __forceinline__ void tdm_load_chunk(const void* gsrc, unsigned lds_addr) {
    unsigned long long ga = (unsigned long long)gsrc;
    v4u g0 = { 1u,                                   // count=1 (valid user D#)
               lds_addr,                             // LDS byte address
               (unsigned)ga,                         // global_addr[31:0]
               (unsigned)((ga >> 32) & 0x1FFFFFFu) | (2u << 30) };  // addr[56:32] | type=2
    v8i g1 = { (int)(2u << 16),        // w0: data_size=2 (4 bytes)
               (int)(4096u << 16),     // w1: tensor_dim0[15:0]=4096
               0,                      // w2: tensor_dim0[31:16]=0, tensor_dim1=0
               (int)(4096u << 16),     // w3: tile_dim0=4096
               0,                      // w4: tile_dim1=0 (unused), tile_dim2=0
               4096,                   // w5: tensor_dim0_stride[31:0]
               0, 0 };                 // w6,w7
    v4i z4 = { 0, 0, 0, 0 };
    v8i z8 = { 0, 0, 0, 0, 0, 0, 0, 0 };
    __builtin_amdgcn_tensor_load_to_lds(g0, g1, z4, z4, z8, 0);
}
#endif

// ---------------------------------------------------------------------------
// Pack the 4 recurrent weight matrices (fp32, K x N row-major) into bf16 WMMA
// B-operand tiles.  Tile order ((nt*32+kt)*4+g) so one workgroup's 4-k-step
// chunk (4 kt x 4 gates = 16KB) is contiguous for the TDM.
// B 32x16 lane layout: col N = lane%16; dword j holds K = 2j+e + 16*(lane/16)
// ---------------------------------------------------------------------------
__global__ __launch_bounds__(256) void pack_w_kernel(
    const float* __restrict__ gh, const float* __restrict__ ih,
    const float* __restrict__ fh, const float* __restrict__ oh,
    unsigned short* __restrict__ wp)
{
    int p = blockIdx.x * blockDim.x + threadIdx.x;
    if (p >= WPACK_ELEMS) return;
    int tile = p >> 9;          // 512 elems per tile
    int r    = p & 511;
    int lane = r >> 4;
    int idx  = r & 15;
    int j = idx >> 1, e = idx & 1;
    int g  = tile & 3;
    int kt = (tile >> 2) & (NKT - 1);
    int nt = tile >> 7;
    int k = kt * KTILE + 2 * j + e + 16 * (lane >> 4);
    int n = nt * 16 + (lane & 15);
    const float* src = (g == 0) ? gh : (g == 1) ? ih : (g == 2) ? fh : oh;
    wp[p] = f2bf(src[k * HH + n]);
}

__global__ __launch_bounds__(256) void init_state_kernel(
    unsigned short* __restrict__ hbf0, float* __restrict__ cbuf)
{
    int idx = blockIdx.x * blockDim.x + threadIdx.x;
    if (idx < BB * HH) { hbf0[idx] = 0; cbuf[idx] = 0.0f; }
}

// ---------------------------------------------------------------------------
// One LSTM timestep.  Grid: 128 blocks of 256 threads (8 waves).
// Block (mgroup, nt): wave w owns M-tile mt = mgroup*8 + w; all 8 waves share
// the B tiles of N-strip nt, staged into LDS by the Tensor Data Mover
// (double-buffered 16KB chunks, TENSORcnt + barriers for sync).
// ---------------------------------------------------------------------------
__global__ __launch_bounds__(256) void lstm_step_kernel(
    const unsigned short* __restrict__ wp,
    const float* __restrict__ x,
    const float* __restrict__ gx, const float* __restrict__ gb,
    const float* __restrict__ ix, const float* __restrict__ ib,
    const float* __restrict__ fx, const float* __restrict__ fb,
    const float* __restrict__ ox, const float* __restrict__ ob,
    const unsigned short* __restrict__ hin,
    unsigned short* __restrict__ hout,
    float* __restrict__ cbuf,
    float* __restrict__ hf,
    int t)
{
    __shared__ __align__(32) unsigned short smem[2 * CHUNK_ELEMS];   // 32KB

    const int lane  = threadIdx.x & 31;
    const int wave  = threadIdx.x >> 5;          // 0..7
    const int nt     = blockIdx.x & 63;
    const int mgroup = blockIdx.x >> 6;          // 0..1
    const int mt     = mgroup * 8 + wave;
    const int lrow  = lane & 15;
    const int lhalf = lane >> 4;

    const unsigned short* wchunk0 = wp + (size_t)(nt * NKT * 4) * 512;  // this nt's strip

    v8f acc0 = {}, acc1 = {}, acc2 = {}, acc3 = {};
    const int arow = (mt * 16 + lrow) * HH;
    const int loff = lane << 4;                  // lane*16 elems within a tile

#if HAVE_TDM
    if (wave == 0)
        tdm_load_chunk(wchunk0, (unsigned)(unsigned long long)&smem[0]);
#else
    {   // cooperative fallback: all threads copy chunk 0
        const uint4* src = (const uint4*)wchunk0;
        uint4* dst = (uint4*)&smem[0];
        for (int i = threadIdx.x; i < CHUNK_BYTES / 16; i += 256) dst[i] = src[i];
    }
#endif

    for (int c = 0; c < NCHUNK; ++c) {
        const int buf = c & 1;
#if HAVE_TDM
        if (wave == 0) {
            if (c + 1 < NCHUNK) {
                tdm_load_chunk(wchunk0 + (size_t)(c + 1) * CHUNK_ELEMS,
                               (unsigned)(unsigned long long)&smem[(buf ^ 1) * CHUNK_ELEMS]);
                __builtin_amdgcn_s_wait_tensorcnt(1);   // chunk c resident (in-order)
            } else {
                __builtin_amdgcn_s_wait_tensorcnt(0);
            }
        }
#else
        if (c + 1 < NCHUNK) {
            const uint4* src = (const uint4*)(wchunk0 + (size_t)(c + 1) * CHUNK_ELEMS);
            uint4* dst = (uint4*)&smem[(buf ^ 1) * CHUNK_ELEMS];
            for (int i = threadIdx.x; i < CHUNK_BYTES / 16; i += 256) dst[i] = src[i];
        }
#endif
        __syncthreads();     // chunk c visible to all waves

        const unsigned short* bbase = &smem[buf * CHUNK_ELEMS];
        #pragma unroll
        for (int kk = 0; kk < CHUNK_KT; ++kk) {
            int kt = c * CHUNK_KT + kk;
            // ----- A operand: 16x32 bf16 tile of h (k = 16*(lane/16) + elem, same order as B) -----
            union { unsigned int u[8]; v16bf v; } A;
            int kbase = kt * KTILE + 16 * lhalf;
            #pragma unroll
            for (int j = 0; j < 8; ++j) {
                int k = kbase + 2 * j;
                A.u[j] = *(const unsigned int*)(hin + arow + k);
            }
            // ----- B operands from LDS: one 32x16 tile per gate -----
            const unsigned short* kb = bbase + (kk * 4) * 512;
            v16bf b0 = *(const v16bf*)(kb + 0 * 512 + loff);
            v16bf b1 = *(const v16bf*)(kb + 1 * 512 + loff);
            v16bf b2 = *(const v16bf*)(kb + 2 * 512 + loff);
            v16bf b3 = *(const v16bf*)(kb + 3 * 512 + loff);

            acc0 = __builtin_amdgcn_wmma_f32_16x16x32_bf16(false, A.v, false, b0,
                                                           (short)0, acc0, false, false);
            acc1 = __builtin_amdgcn_wmma_f32_16x16x32_bf16(false, A.v, false, b1,
                                                           (short)0, acc1, false, false);
            acc2 = __builtin_amdgcn_wmma_f32_16x16x32_bf16(false, A.v, false, b2,
                                                           (short)0, acc2, false, false);
            acc3 = __builtin_amdgcn_wmma_f32_16x16x32_bf16(false, A.v, false, b3,
                                                           (short)0, acc3, false, false);
        }
        __syncthreads();     // everyone done with buf before it is re-filled
    }

    // ----- fused epilogue: x contribution, bias, activations, c/h update -----
    int n = nt * 16 + lrow;                 // C/D layout: N = lane%16
    float w0 = gx[n], w1 = ix[n], w2 = fx[n], w3 = ox[n];
    float c0 = gb[n], c1 = ib[n], c2 = fb[n], c3 = ob[n];

    #pragma unroll
    for (int v = 0; v < 8; ++v) {
        int brow = mt * 16 + v + 8 * lhalf; // C/D layout: M = vgpr + 8*(lane/16)
        float xv = x[brow * TT + t];
        float pg = acc0[v] + xv * w0 + c0;
        float pi = acc1[v] + xv * w1 + c1;
        float pf = acc2[v] + xv * w2 + c2;
        float po = acc3[v] + xv * w3 + c3;
        float gg = TANHF(pg);
        float ii = sigf(pi);
        float ff = sigf(pf);
        float oo = sigf(po);
        int addr = brow * HH + n;
        float cn = gg * ii + cbuf[addr] * ff;
        cbuf[addr] = cn;
        float hn = TANHF(cn) * oo;
        hf[addr]   = hn;
        hout[addr] = f2bf(hn);
    }
}

// final projection: out = h @ ph + pb   (256x10, tiny)
__global__ __launch_bounds__(256) void proj_kernel(
    const float* __restrict__ hf, const float* __restrict__ ph,
    const float* __restrict__ pb, float* __restrict__ out)
{
    int idx = blockIdx.x * blockDim.x + threadIdx.x;
    if (idx >= BB * CC) return;
    int b = idx / CC, c = idx % CC;
    float s = pb[c];
    const float* hr = hf + b * HH;
    for (int k = 0; k < HH; ++k) s += hr[k] * ph[k * CC + c];
    out[idx] = s;
}

extern "C" void kernel_launch(void* const* d_in, const int* in_sizes, int n_in,
                              void* d_out, int out_size, void* d_ws, size_t ws_size,
                              hipStream_t stream) {
    const float* x  = (const float*)d_in[0];
    const float* gx = (const float*)d_in[1];
    const float* gh = (const float*)d_in[2];
    const float* gb = (const float*)d_in[3];
    const float* ix = (const float*)d_in[4];
    const float* ih = (const float*)d_in[5];
    const float* ib = (const float*)d_in[6];
    const float* fx = (const float*)d_in[7];
    const float* fh = (const float*)d_in[8];
    const float* fb = (const float*)d_in[9];
    const float* ox = (const float*)d_in[10];
    const float* oh = (const float*)d_in[11];
    const float* ob = (const float*)d_in[12];
    const float* ph = (const float*)d_in[13];
    const float* pb = (const float*)d_in[14];

    unsigned short* wp   = (unsigned short*)d_ws;
    unsigned short* hbf0 = (unsigned short*)((char*)d_ws + OFF_HBF);
    unsigned short* hbf1 = hbf0 + (size_t)BB * HH;
    float* cbuf = (float*)((char*)d_ws + OFF_C);
    float* hf   = (float*)((char*)d_ws + OFF_HF);

    pack_w_kernel<<<WPACK_ELEMS / 256, 256, 0, stream>>>(gh, ih, fh, oh, wp);
    init_state_kernel<<<(BB * HH) / 256, 256, 0, stream>>>(hbf0, cbuf);

    for (int t = 0; t < TT; ++t) {
        unsigned short* hin  = (t & 1) ? hbf1 : hbf0;
        unsigned short* hout = (t & 1) ? hbf0 : hbf1;
        lstm_step_kernel<<<128, 256, 0, stream>>>(
            wp, x, gx, gb, ix, ib, fx, fb, ox, ob, hin, hout, cbuf, hf, t);
    }

    proj_kernel<<<(BB * CC + 255) / 256, 256, 0, stream>>>(hf, ph, pb, (float*)d_out);
}
